// Cross_Attention_n_30528627540442
// MI455X (gfx1250) — compile-verified, matrix operand order checked by reading, not run on hardware
//
#include <hip/hip_runtime.h>

// ---------------------------------------------------------------------------
// Cross-attention, MI455X (gfx1250, wave32, WMMA bf16, async LDS staging).
//   qp = q@Wq^T + pe ; kp = k@Wk^T + pe ; vp = v@Wv^T + pe   (bf16 in ws)
//   attn = softmax(qp kp^T / 16) ; out = attn vp
//   result = gelu(out Wo^T + bo) + q
// Needs ws_size >= (3*1024*64*256 + 4*256*256 + 64*256) * 2 bytes (~96.3 MB).
// ---------------------------------------------------------------------------

typedef __attribute__((ext_vector_type(16))) __bf16    v16bf;
typedef __attribute__((ext_vector_type(16))) float     v16f;
typedef __attribute__((ext_vector_type(8)))  float     v8f;
typedef __attribute__((ext_vector_type(4)))  unsigned  v4u;
typedef int v4i_t __attribute__((vector_size(16)));    // matches builtin param

#define DIM   256
#define SAMP  64
#define NVIEW 1024

#if __has_builtin(__builtin_amdgcn_global_load_async_to_lds_b128)
#define HAVE_ASYNC_LDS 1
#endif

__device__ __forceinline__ unsigned short f2bf(float f) {
  unsigned u = __builtin_bit_cast(unsigned, f);
  return (unsigned short)((u + 0x7FFFu + ((u >> 16) & 1u)) >> 16);  // RNE
}

__device__ __forceinline__ float bf2f(unsigned short h) {
  return __builtin_bit_cast(float, (unsigned)h << 16);
}

__device__ __forceinline__ v16bf cvt_bf16(v16f x) {
  return __builtin_convertvector(x, v16bf);
}

__device__ __forceinline__ v8f wmma_bf16(v16bf a, v16bf b, v8f c) {
  return __builtin_amdgcn_wmma_f32_16x16x32_bf16(
      /*neg_a=*/false, a, /*neg_b=*/false, b,
      /*c_mod=*/(short)0, c, /*reuse_a=*/false, /*reuse_b=*/false);
}

// Cooperative global->LDS staging in 16B chunks. Uses the gfx1250 async
// LDS-load path (ASYNCcnt) when the builtin exists, else vector copies.
// Caller must __syncthreads() afterwards.
__device__ __forceinline__ void stage_to_lds(void* lds_dst, const void* gsrc,
                                             int bytes, int tid, int nthr) {
#ifdef HAVE_ASYNC_LDS
  auto lp = (__attribute__((address_space(3))) char*)lds_dst;
  auto gp = (__attribute__((address_space(1))) char*)gsrc;   // drops const
  for (int ofs = tid * 16; ofs < bytes; ofs += nthr * 16)
    __builtin_amdgcn_global_load_async_to_lds_b128(
        (__attribute__((address_space(1))) v4i_t*)(gp + ofs),
        (__attribute__((address_space(3))) v4i_t*)(lp + ofs),
        0, 0);
#if __has_builtin(__builtin_amdgcn_s_wait_asynccnt)
  __builtin_amdgcn_s_wait_asynccnt(0);
#else
  asm volatile("s_wait_asynccnt 0x0" ::: "memory");
#endif
#else
  for (int ofs = tid * 16; ofs < bytes; ofs += nthr * 16)
    *(v4u*)((char*)lds_dst + ofs) = *(const v4u*)((const char*)gsrc + ofs);
#endif
}

// ---------------------------------------------------------------------------
// Kernel 0 (prep): bf16-convert the four 256x256 weight matrices AND build
// the bf16 sinusoidal pos-enc table [64][256]. One thread per element.
// wdst layout: [Wq | Wk | Wv | Wo]; pe appended after.
// ---------------------------------------------------------------------------
__global__ __launch_bounds__(256) void ca_prep_kernel(
    const float* __restrict__ Wq, const float* __restrict__ Wk,
    const float* __restrict__ Wv, const float* __restrict__ Wo,
    unsigned short* __restrict__ wdst, unsigned short* __restrict__ pedst)
{
  const int i = blockIdx.x * 256 + threadIdx.x;   // grid = 1088 blocks
  if (i < 4 * DIM * DIM) {
    const int which = i >> 16;
    const float* src = (which == 0) ? Wq : (which == 1) ? Wk
                     : (which == 2) ? Wv : Wo;
    wdst[i] = f2bf(src[i & 0xFFFF]);
  } else if (i < 4 * DIM * DIM + SAMP * DIM) {
    const int idx = i - 4 * DIM * DIM;
    const int s = idx >> 8, d = idx & 255;
    const float freq = __powf(10000.0f, -(float)(2 * (d >> 1)) * (1.0f / 256.0f));
    const float ang  = (float)s * freq;
    pedst[idx] = f2bf((d & 1) ? __cosf(ang) : __sinf(ang));
  }
}

// ---------------------------------------------------------------------------
// Kernel 1: projection GEMM + positional encoding, f32 -> bf16.
// grid.x = 512 (128-row tiles), grid.y = 3 (q/k/v). 256 threads = 8 waves;
// weight matrix + pe table (bf16) staged once in LDS, reused by all 8 waves.
// vp (proj==2) is stored transposed within each ray block: [b][D][S].
// ---------------------------------------------------------------------------
__global__ __launch_bounds__(256) void ca_proj_kernel(
    const float* __restrict__ q, const float* __restrict__ k,
    const float* __restrict__ v,
    const unsigned short* __restrict__ wmat,   // bf16 [4][256*256]
    const unsigned short* __restrict__ pe,     // bf16 [64][256]
    unsigned short* __restrict__ oq, unsigned short* __restrict__ ok,
    unsigned short* __restrict__ ov)
{
  __shared__ __align__(32) unsigned short s_w[DIM * DIM];    // 128 KB
  __shared__ __align__(32) unsigned short s_pe[SAMP * DIM];  //  32 KB

  const int proj = blockIdx.y;
  const float* X = (proj == 0) ? q : (proj == 1) ? k : v;
  unsigned short* O = (proj == 0) ? oq : (proj == 1) ? ok : ov;

  stage_to_lds(s_w, wmat + (size_t)proj * DIM * DIM, DIM * DIM * 2,
               threadIdx.x, 256);
  stage_to_lds(s_pe, pe, SAMP * DIM * 2, threadIdx.x, 256);
  __syncthreads();

  const int wave = threadIdx.x >> 5;
  const int lane = threadIdx.x & 31;
  const int ln   = lane & 15;   // row/col within 16-wide half
  const int hi   = lane >> 4;   // which K half of the fragment
  const int m0   = blockIdx.x * 128 + wave * 16;

  const float* xrow = X + (size_t)(m0 + ln) * DIM;    // A: one row per lane

  for (int nt = 0; nt < 16; ++nt) {
    const int n0 = nt * 16;
    const unsigned short* wrow = s_w + (n0 + ln) * DIM;  // B = W^T (W rows)
    v8f acc = {};
    #pragma unroll
    for (int ks = 0; ks < 8; ++ks) {
      const int off = ks * 32 + hi * 16;
      __builtin_prefetch(xrow + off + 64, 0, 1);
      v16bf a = cvt_bf16(*(const v16f*)(xrow + off));
      v16bf b = *(const v16bf*)(wrow + off);
      acc = wmma_bf16(a, b, acc);
    }
    // C layout: VGPR r, lane -> row m0 + r + 8*hi, col n0 + ln
    const int dcol = n0 + ln;
    #pragma unroll
    for (int r = 0; r < 8; ++r) {
      const int row = m0 + r + 8 * hi;
      const int s   = row & (SAMP - 1);
      const float val = acc[r] + bf2f(s_pe[s * DIM + dcol]);
      if (proj == 2) {
        const int blk = row >> 6;                       // ray block
        O[(size_t)blk * (DIM * SAMP) + dcol * SAMP + (row & 63)] = f2bf(val);
      } else {
        O[(size_t)row * DIM + dcol] = f2bf(val);
      }
    }
  }
}

// ---------------------------------------------------------------------------
// Kernel 2: fused attention + output projection per ray block.
// 1024 blocks, 128 threads = 4 waves; kb/vb staged in LDS once per block.
// ---------------------------------------------------------------------------
__global__ __launch_bounds__(128) void ca_attn_kernel(
    const unsigned short* __restrict__ qp, const unsigned short* __restrict__ kp,
    const unsigned short* __restrict__ vpT,
    const unsigned short* __restrict__ WoBf,           // bf16 [256*256]
    const float* __restrict__ bo,
    const float* __restrict__ resid, float* __restrict__ out)
{
  __shared__ __align__(32) unsigned short s_k[SAMP * DIM];      // 32 KB
  __shared__ __align__(32) unsigned short s_v[DIM * SAMP];      // 32 KB ([D][S])
  __shared__ __align__(32) float          s_scores[SAMP * SAMP];// 16 KB
  __shared__ __align__(32) unsigned short s_attn[SAMP * SAMP];  //  8 KB
  __shared__ __align__(32) unsigned short s_out[SAMP * DIM];    // 32 KB

  const int blk  = blockIdx.x;
  const int lane = threadIdx.x & 31;
  const int ln   = lane & 15;
  const int hi   = lane >> 4;
  const int m0   = (threadIdx.x >> 5) * 16;

  const unsigned short* qb = qp + (size_t)blk * SAMP * DIM;

  stage_to_lds(s_k, kp  + (size_t)blk * SAMP * DIM, SAMP * DIM * 2,
               threadIdx.x, 128);
  stage_to_lds(s_v, vpT + (size_t)blk * DIM * SAMP, DIM * SAMP * 2,
               threadIdx.x, 128);
  __syncthreads();

  // -------- Phase A: scores = (qp kp^T) / sqrt(D) --------
  {
    v8f acc[4] = {v8f{}, v8f{}, v8f{}, v8f{}};
    #pragma unroll
    for (int ks = 0; ks < 8; ++ks) {
      const int off = ks * 32 + hi * 16;
      v16bf a = *(const v16bf*)(qb + (size_t)(m0 + ln) * DIM + off);
      #pragma unroll
      for (int nt = 0; nt < 4; ++nt) {
        v16bf bf = *(const v16bf*)(s_k + (nt * 16 + ln) * DIM + off);
        acc[nt] = wmma_bf16(a, bf, acc[nt]);
      }
    }
    const float scale = 0.0625f;   // 1/sqrt(256)
    #pragma unroll
    for (int nt = 0; nt < 4; ++nt)
      #pragma unroll
      for (int r = 0; r < 8; ++r)
        s_scores[(m0 + r + 8 * hi) * SAMP + nt * 16 + ln] = acc[nt][r] * scale;
  }
  __syncthreads();

  // -------- Phase B: row softmax -> bf16 --------
  if (threadIdx.x < SAMP) {
    float* row = s_scores + threadIdx.x * SAMP;
    float mx = -1e30f;
    for (int j = 0; j < SAMP; ++j) mx = fmaxf(mx, row[j]);
    float sum = 0.f;
    for (int j = 0; j < SAMP; ++j) { float e = __expf(row[j] - mx); row[j] = e; sum += e; }
    const float rs = 1.0f / sum;
    for (int j = 0; j < SAMP; ++j) s_attn[threadIdx.x * SAMP + j] = f2bf(row[j] * rs);
  }
  __syncthreads();

  // -------- Phase C: out = attn @ vp (vp transposed -> contiguous B) --------
  for (int nt = 0; nt < 16; ++nt) {
    const int n0 = nt * 16;
    v8f acc = {};
    #pragma unroll
    for (int ks = 0; ks < 2; ++ks) {
      const int off = ks * 32 + hi * 16;
      v16bf a  = *(const v16bf*)(s_attn + (m0 + ln) * SAMP + off);
      v16bf bf = *(const v16bf*)(s_v + (n0 + ln) * SAMP + off);
      acc = wmma_bf16(a, bf, acc);
    }
    #pragma unroll
    for (int r = 0; r < 8; ++r)
      s_out[(m0 + r + 8 * hi) * DIM + n0 + ln] = f2bf(acc[r]);
  }
  __syncthreads();

  // -------- Phase D: gelu(out Wo^T + bo) + residual --------
  for (int nt = 0; nt < 16; ++nt) {
    const int n0   = nt * 16;
    const int dcol = n0 + ln;
    v8f acc = {};
    #pragma unroll
    for (int ks = 0; ks < 8; ++ks) {
      const int off = ks * 32 + hi * 16;
      v16bf a  = *(const v16bf*)(s_out + (m0 + ln) * DIM + off);
      v16bf bf = *(const v16bf*)(WoBf + (size_t)dcol * DIM + off);  // L2 hit
      acc = wmma_bf16(a, bf, acc);
    }
    const float bias = bo[dcol];
    #pragma unroll
    for (int r = 0; r < 8; ++r) {
      const int row = m0 + r + 8 * hi;
      const size_t gi = ((size_t)blk * SAMP + row) * DIM + dcol;
      const float x = acc[r] + bias;
      const float g = 0.5f * x * (1.0f + erff(x * 0.70710678118f));  // exact gelu
      out[gi] = g + resid[gi];
    }
  }
}

// ---------------------------------------------------------------------------
extern "C" void kernel_launch(void* const* d_in, const int* in_sizes, int n_in,
                              void* d_out, int out_size, void* d_ws, size_t ws_size,
                              hipStream_t stream) {
  const float* q  = (const float*)d_in[0];
  const float* k  = (const float*)d_in[1];
  const float* v  = (const float*)d_in[2];
  const float* Wq = (const float*)d_in[3];
  const float* Wk = (const float*)d_in[4];
  const float* Wv = (const float*)d_in[5];
  const float* Wo = (const float*)d_in[6];
  const float* bo = (const float*)d_in[7];
  float* out = (float*)d_out;

  const size_t nElem = (size_t)NVIEW * SAMP * DIM;   // 16.78M
  unsigned short* wq   = (unsigned short*)d_ws;
  unsigned short* wk   = wq + nElem;
  unsigned short* wv   = wk + nElem;
  unsigned short* wmat = wv + nElem;                 // 4 * 65536 bf16
  unsigned short* woBf = wmat + 3 * (size_t)DIM * DIM;
  unsigned short* peT  = wmat + 4 * (size_t)DIM * DIM;  // 64*256 bf16

  ca_prep_kernel<<<dim3(1088), 256, 0, stream>>>(Wq, Wk, Wv, Wo, wmat, peT);
  ca_proj_kernel<<<dim3(512, 3), 256, 0, stream>>>(q, k, v, wmat, peT,
                                                   wq, wk, wv);
  ca_attn_kernel<<<dim3(NVIEW), 128, 0, stream>>>(wq, wk, wv, woBf, bo, q, out);
}
